// Biaffine_10007273799977
// MI455X (gfx1250) — matrix operand order repaired and checked by reading, not compiled, not run on hardware
//
#include <hip/hip_runtime.h>
#include <stdint.h>

// Biaffine: out[b,x,y] = (Daug@U)[b,x,:] . H[b,y,:] + dlin[b,x] + hlin[b,y]
// bf16 WMMA (16x16x32) two-stage GEMM, fp32 accumulate.
// GEMM2: A-tile via TDM tensor_load_to_lds (TENSORcnt, HW padding),
//        B-tile via global_load_async_to_lds_b128 (ASYNCcnt).

#define B_DIM 32
#define S_DIM 512
#define DD    1024
#define NROWS (B_DIM * S_DIM)   // 16384

#define BM 128
#define BN 128
#define BK 32
#define LDK 40                  // padded LDS row stride (elements) -> conflict-free b128 frag loads
#define NKT (DD / BK)           // 32

typedef __attribute__((ext_vector_type(16))) __bf16 v16bf;
typedef __attribute__((ext_vector_type(8)))  float  v8f;
typedef int v4i __attribute__((vector_size(16)));          // payload type of the async b128 builtin
typedef __attribute__((address_space(1))) v4i g_v4i;       // global
typedef __attribute__((address_space(3))) v4i l_v4i;       // LDS
typedef __attribute__((address_space(3))) void lds_void;

typedef unsigned int u32x4 __attribute__((ext_vector_type(4)));
typedef int          i32x4 __attribute__((ext_vector_type(4)));
typedef int          i32x8 __attribute__((ext_vector_type(8)));

union FragBF { v16bf v; uint4 q[2]; };
union Pack8  { __bf16 h[8]; uint4 q; };

// ---- gfx1250 async-to-LDS copy (ASYNCcnt) --------------------------------
#if __has_builtin(__builtin_amdgcn_global_load_async_to_lds_b128)
#define HAVE_ASYNC_LDS 1
#else
#define HAVE_ASYNC_LDS 0
#endif

// ---- gfx1250 Tensor Data Mover -------------------------------------------
#if __has_builtin(__builtin_amdgcn_tensor_load_to_lds)
#define HAVE_TDM 1
#else
#define HAVE_TDM 0
#endif

__device__ __forceinline__ void async_copy_b128(const void* g, void* l) {
#if HAVE_ASYNC_LDS
  __builtin_amdgcn_global_load_async_to_lds_b128(
      (g_v4i*)(void*)g, (l_v4i*)l, 0, 0);
#else
  *(uint4*)l = *(const uint4*)g;
#endif
}

__device__ __forceinline__ void wait_async() {
#if __has_builtin(__builtin_amdgcn_s_wait_asynccnt)
  __builtin_amdgcn_s_wait_asynccnt(0);
#elif HAVE_ASYNC_LDS
  asm volatile("s_wait_asynccnt 0x0" ::: "memory");
#endif
}

__device__ __forceinline__ void wait_tensor() {
#if __has_builtin(__builtin_amdgcn_s_wait_tensorcnt)
  __builtin_amdgcn_s_wait_tensorcnt(0);
#elif HAVE_TDM
  asm volatile("s_wait_tensorcnt 0x0" ::: "memory");
#endif
}

// TDM: load 2D tile [128 rows x 32 bf16], global row stride 1024 elems,
// into LDS with HW padding: +16B after every 64B row => 80B rows (LDK=40).
__device__ __forceinline__ void tdm_load_tile_bf16(const __bf16* gsrc, void* ldst) {
#if HAVE_TDM
  const unsigned long long ga = (unsigned long long)(uintptr_t)gsrc;
  const unsigned int la = (unsigned int)(uintptr_t)(lds_void*)ldst;
  u32x4 g0;
  g0[0] = 1u;                                    // count=1 (valid), is_restore=0, gather off
  g0[1] = la;                                    // lds_addr [63:32]
  g0[2] = (unsigned int)ga;                      // global_addr [95:64]
  g0[3] = (unsigned int)(ga >> 32) | (2u << 30); // global_addr[56:32] | type=2 ("image")
  i32x8 g1;
  g1[0] = (1 << 16)        // data_size = 1 -> 2-byte elements
        | (1 << 20)        // pad_enable
        | (3 << 22)        // pad_interval: every 16 DWORDs (64B)
        | (3 << 25);       // pad_amount: 4 DWORDs (16B)
  g1[1] = (int)(1024u << 16);  // tensor_dim0 = 1024 (elems), low 16 bits
  g1[2] = (int)(512u << 16);   // tensor_dim0 hi = 0 | tensor_dim1 = 512, low 16
  g1[3] = (int)(32u << 16);    // tensor_dim1 hi = 0 | tile_dim0 = 32
  g1[4] = 128;                 // tile_dim1 = 128, tile_dim2 = 0 (unused)
  g1[5] = 1024;                // tensor_dim0_stride = 1024 (elems), low 32
  g1[6] = 0;                   // stride0 hi | tensor_dim1_stride low (unused)
  g1[7] = 0;
  const i32x4 zz4 = {0, 0, 0, 0};
  const i32x8 zz8 = {0, 0, 0, 0, 0, 0, 0, 0};
  __builtin_amdgcn_tensor_load_to_lds(g0, g1, zz4, zz4, zz8, 0);
#else
  (void)gsrc; (void)ldst;
#endif
}

// A fragment (16x32, MxK): lane m=lane&15, h=lane>>4. VGPR0..3 = K h*8..h*8+7, VGPR4..7 = K 16+h*8..
__device__ __forceinline__ v16bf ldfragA(const __bf16* rowptr, int h) {
  FragBF f;
  f.q[0] = *(const uint4*)(rowptr + h * 8);
  f.q[1] = *(const uint4*)(rowptr + 16 + h * 8);
  return f.v;
}
// B fragment (32x16, KxN), LDS holds B^T rows [n][k]: lane n=lane&15, h=lane>>4. VGPR0..3 = K h*8..h*8+7, VGPR4..7 = K 16+h*8..
__device__ __forceinline__ v16bf ldfragB(const __bf16* rowptr, int h) {
  FragBF f;
  f.q[0] = *(const uint4*)(rowptr + h * 8);
  f.q[1] = *(const uint4*)(rowptr + 16 + h * 8);
  return f.v;
}

// ---------------------------------------------------------------------------
// prep: dlin[r] = D[r,:].W[0:1024] + W[1024]; hlin[r] = H[r,:].W[1025:2049];
//       Hb = bf16(H). One wave per row.
// ---------------------------------------------------------------------------
__global__ __launch_bounds__(256) void prep_kernel(
    const float* __restrict__ D, const float* __restrict__ H,
    const float* __restrict__ W, __bf16* __restrict__ Hb,
    float* __restrict__ dlin, float* __restrict__ hlin) {
  const int w = threadIdx.x >> 5, lane = threadIdx.x & 31;
  const int row = blockIdx.x * 8 + w;                 // 0..16383
  const float* drow = D + (size_t)row * DD;
  const float* hrow = H + (size_t)row * DD;
  __bf16* hbrow = Hb + (size_t)row * DD;
  float ds = 0.f, hs = 0.f;
#pragma unroll 4
  for (int i = 0; i < DD / 32; ++i) {
    const int k = i * 32 + lane;
    const float dv = drow[k];
    const float hv = hrow[k];
    ds += dv * W[k];
    hs += hv * W[DD + 1 + k];
    hbrow[k] = (__bf16)hv;
  }
#pragma unroll
  for (int off = 16; off > 0; off >>= 1) {
    ds += __shfl_xor(ds, off, 32);
    hs += __shfl_xor(hs, off, 32);
  }
  if (lane == 0) { dlin[row] = ds + W[DD]; hlin[row] = hs; }
}

// ---------------------------------------------------------------------------
// GEMM1: T[16384,1024] = D[16384,1024] @ U[0:1024,1024] + U[1024,:], bf16 out
// (fp32 sources -> VALU cvt staging with register prefetch pipeline)
// ---------------------------------------------------------------------------
__global__ __launch_bounds__(256) void gemm1_kernel(
    const float* __restrict__ D, const float* __restrict__ U,
    __bf16* __restrict__ Tb) {
  __shared__ alignas(16) __bf16 sA[2][BM * LDK];
  __shared__ alignas(16) __bf16 sB[2][BN * LDK];
  const int t = threadIdx.x;
  const int mt = blockIdx.x >> 3, nt = blockIdx.x & 7;
  const int row0 = mt * BM, col0 = nt * BN;
  const int lane = t & 31, w = t >> 5;
  const int wm = w >> 1, wn = w & 1;                  // 4x2 wave grid
  const int fm = lane & 15, fh = lane >> 4;
  const int arow = t >> 2, aseg = t & 3;              // A staging: 2 chunks of 8 elems
  const int bn = t & 127, bk0 = (t >> 7) * 16;        // B staging: 16 elems along K (transpose U)

  float fa[2][8];
  float fb[16];

  auto fetch = [&](int k0) {
#pragma unroll
    for (int i = 0; i < 2; ++i) {
      const float* g = D + (size_t)(row0 + arow + i * 64) * DD + k0 + aseg * 8;
      const float4 x = *(const float4*)g;
      const float4 y = *(const float4*)(g + 4);
      fa[i][0] = x.x; fa[i][1] = x.y; fa[i][2] = x.z; fa[i][3] = x.w;
      fa[i][4] = y.x; fa[i][5] = y.y; fa[i][6] = y.z; fa[i][7] = y.w;
    }
#pragma unroll
    for (int kk = 0; kk < 16; ++kk)
      fb[kk] = U[(size_t)(k0 + bk0 + kk) * DD + col0 + bn];
  };
  auto store_stage = [&](int buf) {
#pragma unroll
    for (int i = 0; i < 2; ++i) {
      Pack8 p;
#pragma unroll
      for (int e = 0; e < 8; ++e) p.h[e] = (__bf16)fa[i][e];
      *(uint4*)&sA[buf][(arow + i * 64) * LDK + aseg * 8] = p.q;
    }
    Pack8 p0, p1;
#pragma unroll
    for (int e = 0; e < 8; ++e) { p0.h[e] = (__bf16)fb[e]; p1.h[e] = (__bf16)fb[8 + e]; }
    *(uint4*)&sB[buf][bn * LDK + bk0] = p0.q;
    *(uint4*)&sB[buf][bn * LDK + bk0 + 8] = p1.q;
  };

  const v8f vzero = {0.f, 0.f, 0.f, 0.f, 0.f, 0.f, 0.f, 0.f};
  v8f acc[2][4];
#pragma unroll
  for (int i = 0; i < 2; ++i)
#pragma unroll
    for (int j = 0; j < 4; ++j) acc[i][j] = vzero;

  fetch(0);
  store_stage(0);
  __syncthreads();

  for (int kt = 0; kt < NKT; ++kt) {
    const int cur = kt & 1;
    if (kt + 1 < NKT) fetch((kt + 1) * BK);
    const __bf16* As = sA[cur];
    const __bf16* Bs = sB[cur];
    v16bf af[2], bfr[4];
#pragma unroll
    for (int i = 0; i < 2; ++i)
      af[i] = ldfragA(As + (wm * 32 + i * 16 + fm) * LDK, fh);
#pragma unroll
    for (int j = 0; j < 4; ++j)
      bfr[j] = ldfragB(Bs + (wn * 64 + j * 16 + fm) * LDK, fh);
#pragma unroll
    for (int i = 0; i < 2; ++i)
#pragma unroll
      for (int j = 0; j < 4; ++j)
        acc[i][j] = __builtin_amdgcn_wmma_f32_16x16x32_bf16(
            false, af[i], false, bfr[j], (short)0, acc[i][j], false, false);
    if (kt + 1 < NKT) store_stage(cur ^ 1);
    __syncthreads();
  }

  // epilogue: + bias row U[1024,:], convert to bf16
  const float* Ubias = U + (size_t)DD * DD;
#pragma unroll
  for (int i = 0; i < 2; ++i) {
    const int rbase = row0 + wm * 32 + i * 16 + fh * 8;
#pragma unroll
    for (int j = 0; j < 4; ++j) {
      const int col = col0 + wn * 64 + j * 16 + fm;
      const float bias = Ubias[col];
#pragma unroll
      for (int r = 0; r < 8; ++r)
        Tb[(size_t)(rbase + r) * DD + col] = (__bf16)(acc[i][j][r] + bias);
    }
  }
}

// ---------------------------------------------------------------------------
// GEMM2: out[b,x,y] = Tb[b,x,:] . Hb[b,y,:] + dlin[b,x] + hlin[b,y]
// A-tile: TDM (one DMA per tile, HW padding). B-tile: async-to-LDS copies.
// ---------------------------------------------------------------------------
__global__ __launch_bounds__(256) void gemm2_kernel(
    const __bf16* __restrict__ Tb, const __bf16* __restrict__ Hb,
    const float* __restrict__ dlin, const float* __restrict__ hlin,
    float* __restrict__ out) {
  __shared__ alignas(16) __bf16 sA[2][BM * LDK];
  __shared__ alignas(16) __bf16 sB[2][BN * LDK];
  const int t = threadIdx.x;
  const int b  = blockIdx.x >> 4;
  const int xt = (blockIdx.x >> 2) & 3, yt = blockIdx.x & 3;
  const int x0 = xt * BM, y0 = yt * BN;
  const int lane = t & 31, w = t >> 5;
  const int wm = w >> 1, wn = w & 1;
  const int fm = lane & 15, fh = lane >> 4;
  const int arow = t >> 2, aseg = t & 3;

  const __bf16* Abase = Tb + (size_t)b * S_DIM * DD;
  const __bf16* Bbase = Hb + (size_t)b * S_DIM * DD;

  auto stage = [&](int buf, int k0) {
#if HAVE_TDM
    if (t < 32)   // one wave issues the A-tile DMA descriptor
      tdm_load_tile_bf16(Abase + (size_t)x0 * DD + k0, &sA[buf][0]);
#else
#pragma unroll
    for (int i = 0; i < 2; ++i) {
      const int r = arow + i * 64;
      async_copy_b128(Abase + (size_t)(x0 + r) * DD + k0 + aseg * 8,
                      &sA[buf][r * LDK + aseg * 8]);
    }
#endif
#pragma unroll
    for (int i = 0; i < 2; ++i) {
      const int r = arow + i * 64;
      async_copy_b128(Bbase + (size_t)(y0 + r) * DD + k0 + aseg * 8,
                      &sB[buf][r * LDK + aseg * 8]);
    }
  };

  const v8f vzero = {0.f, 0.f, 0.f, 0.f, 0.f, 0.f, 0.f, 0.f};
  v8f acc[2][4];
#pragma unroll
  for (int i = 0; i < 2; ++i)
#pragma unroll
    for (int j = 0; j < 4; ++j) acc[i][j] = vzero;

  stage(0, 0);
  wait_async();
#if HAVE_TDM
  if (t < 32) wait_tensor();
#endif
  __syncthreads();

  for (int kt = 0; kt < NKT; ++kt) {
    const int cur = kt & 1;
    if (kt + 1 < NKT) stage(cur ^ 1, (kt + 1) * BK);  // DMA next tile
    const __bf16* As = sA[cur];
    const __bf16* Bs = sB[cur];
    v16bf af[2], bfr[4];
#pragma unroll
    for (int i = 0; i < 2; ++i)
      af[i] = ldfragA(As + (wm * 32 + i * 16 + fm) * LDK, fh);
#pragma unroll
    for (int j = 0; j < 4; ++j)
      bfr[j] = ldfragB(Bs + (wn * 64 + j * 16 + fm) * LDK, fh);
#pragma unroll
    for (int i = 0; i < 2; ++i)
#pragma unroll
      for (int j = 0; j < 4; ++j)
        acc[i][j] = __builtin_amdgcn_wmma_f32_16x16x32_bf16(
            false, af[i], false, bfr[j], (short)0, acc[i][j], false, false);
    wait_async();               // own async B copies into nxt done
#if HAVE_TDM
    if (t < 32) wait_tensor();  // A-tile DMA done (issuing wave only)
#endif
    __syncthreads();            // all waves' copies visible
  }

  // epilogue: + dlin[row] + hlin[col]
  float hl[4];
#pragma unroll
  for (int j = 0; j < 4; ++j)
    hl[j] = hlin[b * S_DIM + y0 + wn * 64 + j * 16 + fm];
#pragma unroll
  for (int i = 0; i < 2; ++i) {
    const int rbase = x0 + wm * 32 + i * 16 + fh * 8;
#pragma unroll
    for (int r = 0; r < 8; ++r) {
      const int row = rbase + r;
      const float dl = dlin[b * S_DIM + row];
      float* orow = out + ((size_t)b * S_DIM + row) * S_DIM;
#pragma unroll
      for (int j = 0; j < 4; ++j) {
        const int col = y0 + wn * 64 + j * 16 + fm;
        orow[col] = acc[i][j][r] + dl + hl[j];
      }
    }
  }
}

// ---------------------------------------------------------------------------
extern "C" void kernel_launch(void* const* d_in, const int* in_sizes, int n_in,
                              void* d_out, int out_size, void* d_ws, size_t ws_size,
                              hipStream_t stream) {
  (void)in_sizes; (void)n_in; (void)out_size; (void)ws_size;
  const float* D = (const float*)d_in[0];
  const float* H = (const float*)d_in[1];
  const float* U = (const float*)d_in[2];
  const float* W = (const float*)d_in[3];
  float* out = (float*)d_out;

  // workspace layout
  const size_t TB_BYTES = (size_t)NROWS * DD * 2;     // 32 MiB (bf16 T)
  char* ws = (char*)d_ws;
  __bf16* Tb   = (__bf16*)ws;
  __bf16* Hb   = (__bf16*)(ws + TB_BYTES);
  float*  dlin = (float*)(ws + 2 * TB_BYTES);
  float*  hlin = dlin + NROWS;

  prep_kernel<<<NROWS / 8, 256, 0, stream>>>(D, H, W, Hb, dlin, hlin);
  gemm1_kernel<<<(NROWS / BM) * (DD / BN), 256, 0, stream>>>(D, U, Tb);
  gemm2_kernel<<<B_DIM * (S_DIM / BM) * (S_DIM / BN), 256, 0, stream>>>(Tb, Hb, dlin, hlin, out);
}
